// ScaledDotProductAttention_47167330845500
// MI455X (gfx1250) — matrix operand order repaired and checked by reading, not compiled, not run on hardware
//
#include <hip/hip_runtime.h>

typedef __attribute__((ext_vector_type(16))) _Float16 v16h;
typedef __attribute__((ext_vector_type(4)))  _Float16 v4h;
typedef __attribute__((ext_vector_type(8)))  float    v8f;

#define NH    16
#define SEQ   2048
#define DHEAD 128
#define QTILE 128          // query rows per workgroup (8 waves x 16)
#define KVT   32           // kv columns per step
#define KPAD  136          // 128 + 8 halves  (272B row stride -> bank rotation)
#define VPAD  36           // 32 + 4 halves   (72B row stride)
#define PPAD  36

union H16 { v16h v; v4h q[4]; _Float16 h[16]; };
union H4  { v4h  v; _Float16 h[4]; };

// DPP16 cross-lane mov (VALU, no LDS). bound_ctrl=1, all rows/banks enabled.
template <int CTRL>
__device__ __forceinline__ float dpp_mov(float x) {
    union { float f; int i; } a, b;
    a.f = x;
    b.i = __builtin_amdgcn_update_dpp(0, a.i, CTRL, 0xf, 0xf, true);
    return b.f;
}
// Reduce across each 16-lane half: xor1, xor2 (quad_perm), then quad-uniform
// mirror exchanges (row_half_mirror, row_mirror). All lanes end with the result.
__device__ __forceinline__ float rowmax16(float x) {
    x = fmaxf(x, dpp_mov<0xB1>(x));    // quad_perm [1,0,3,2]  (xor 1)
    x = fmaxf(x, dpp_mov<0x4E>(x));    // quad_perm [2,3,0,1]  (xor 2)
    x = fmaxf(x, dpp_mov<0x141>(x));   // row_half_mirror      (octet exchange)
    x = fmaxf(x, dpp_mov<0x140>(x));   // row_mirror           (16-lane exchange)
    return x;
}
__device__ __forceinline__ float rowsum16(float x) {
    x += dpp_mov<0xB1>(x);
    x += dpp_mov<0x4E>(x);
    x += dpp_mov<0x141>(x);
    x += dpp_mov<0x140>(x);
    return x;
}

__launch_bounds__(256, 1)
__global__ void fa_alibi_kernel(const float* __restrict__ q,
                                const float* __restrict__ k,
                                const float* __restrict__ v,
                                float* __restrict__ out)
{
    // double-buffered K (row-major) and V^T tiles, f16
    __shared__ __align__(16) _Float16 Klds[2][KVT * KPAD];
    __shared__ __align__(16) _Float16 Vt[2][DHEAD * VPAD];
    __shared__ __align__(16) _Float16 Plds[8 * 16 * PPAD];    // per-wave P transpose patch

    const int tid   = threadIdx.x;
    const int wave  = tid >> 5;
    const int lane  = tid & 31;
    const int lhalf = lane >> 4;          // 0: lanes 0-15, 1: lanes 16-31
    const int l16   = lane & 15;
    const int koff  = lhalf ? 8 : 0;      // A-matrix K offset for upper lane half

    const int q0    = blockIdx.x * QTILE;
    const int head  = blockIdx.y;
    const int batch = blockIdx.z;
    const int mbase = q0 + wave * 16;                       // wave's first query row
    const size_t bh = ((size_t)batch * NH + head) * SEQ;

    // exp2-domain constants: softmax uses 2^x so v_exp_f32 needs no pre-multiply
    const float kLog2e = 1.4426950408889634f;
    const float scl2   = 0.08838834764831845f * kLog2e;           // (1/sqrt(128))*log2e
    const float slope2 = exp2f(-0.5f * (float)(head + 1)) * kLog2e;

    // ---- Q A-fragments: 4 frags of 16x32 over D=128, fp32 -> f16 ----
    H16 qf[4];
    {
        const float* qrp = q + (bh + (size_t)(mbase + l16)) * DHEAD;
        #pragma unroll
        for (int f = 0; f < 4; ++f) {
            #pragma unroll
            for (int g = 0; g < 4; ++g) {
                const int kk = g * 4 + 2 * koff; // {0,4,8,12}+16*lhalf
                const float4 x = *(const float4*)(qrp + f * 32 + kk);
                H4 hh;
                hh.h[0] = (_Float16)x.x; hh.h[1] = (_Float16)x.y;
                hh.h[2] = (_Float16)x.z; hh.h[3] = (_Float16)x.w;
                qf[f].q[g] = hh.v;
            }
        }
    }

    v8f   acc[8];            // O accumulator: 16x128 f32 (row = r + 8*lhalf, col = 16f + l16)
    float mrun[8], lrun[8];
    #pragma unroll
    for (int f = 0; f < 8; ++f) acc[f] = (v8f){0.f,0.f,0.f,0.f,0.f,0.f,0.f,0.f};
    #pragma unroll
    for (int r = 0; r < 8; ++r) { mrun[r] = -1e30f; lrun[r] = 0.0f; }

    const int nsteps = (q0 + QTILE) / KVT;                  // causal bound

    // per-thread staging coordinates for the cooperative 32x128 tile copy
    const int srow[4] = { (tid + 0) >> 5, (tid + 256) >> 5, (tid + 512) >> 5, (tid + 768) >> 5 };
    const int scol    = (tid & 31) * 4;   // float index within row (same for all 4 chunks)

    // ---- prologue: stage tile 0 into buffer 0 ----
    {
        #pragma unroll
        for (int it = 0; it < 4; ++it) {
            const size_t gofs = (bh + (size_t)srow[it]) * DHEAD + scol;
            const float4 kx = *(const float4*)(k + gofs);
            const float4 vx = *(const float4*)(v + gofs);
            H4 kh;
            kh.h[0] = (_Float16)kx.x; kh.h[1] = (_Float16)kx.y;
            kh.h[2] = (_Float16)kx.z; kh.h[3] = (_Float16)kx.w;
            *(v4h*)&Klds[0][srow[it] * KPAD + scol] = kh.v;
            Vt[0][(scol + 0) * VPAD + srow[it]] = (_Float16)vx.x;
            Vt[0][(scol + 1) * VPAD + srow[it]] = (_Float16)vx.y;
            Vt[0][(scol + 2) * VPAD + srow[it]] = (_Float16)vx.z;
            Vt[0][(scol + 3) * VPAD + srow[it]] = (_Float16)vx.w;
        }
    }
    __syncthreads();

    for (int step = 0; step < nsteps; ++step) {
        const int cur  = step & 1;
        const int j0   = step * KVT;
        const bool more = (step + 1) < nsteps;

        // ---- issue next tile's global loads early (latency hidden by compute) ----
        float4 kreg[4], vreg[4];
        if (more) {
            #pragma unroll
            for (int it = 0; it < 4; ++it) {
                const size_t gofs = (bh + (size_t)(j0 + KVT + srow[it])) * DHEAD + scol;
                kreg[it] = *(const float4*)(k + gofs);
                vreg[it] = *(const float4*)(v + gofs);
                if (step + 2 < nsteps) {   // warm L2 for step+2 -> global_prefetch_b8
                    __builtin_prefetch(k + gofs + (size_t)KVT * DHEAD, 0, 1);
                    __builtin_prefetch(v + gofs + (size_t)KVT * DHEAD, 0, 1);
                }
            }
        }

        // ---- preload ALL K B-fragments, then 8 back-to-back QK WMMAs ----
        H16 kf[2][4];
        #pragma unroll
        for (int sub = 0; sub < 2; ++sub) {
            const int jj = sub * 16 + l16;               // B col n -> K row j0+jj
            #pragma unroll
            for (int f = 0; f < 4; ++f) {
                const int base = jj * KPAD + f * 32 + (lhalf ? 16 : 0);
                #pragma unroll
                for (int g = 0; g < 4; ++g)
                    kf[sub][f].q[g] = *(const v4h*)&Klds[cur][base + g * 4];
            }
        }
        v8f s[2];
        #pragma unroll
        for (int sub = 0; sub < 2; ++sub) {
            v8f c = (v8f){0.f,0.f,0.f,0.f,0.f,0.f,0.f,0.f};
            #pragma unroll
            for (int f = 0; f < 4; ++f)
                c = __builtin_amdgcn_wmma_f32_16x16x32_f16(
                        false, qf[f].v, false, kf[sub][f].v, (short)0, c, false, false);
            s[sub] = c;
        }

        // ---- online softmax in exp2 domain; ALiBi slope*j (row term cancels) ----
        float sc[2][8];
        #pragma unroll
        for (int sub = 0; sub < 2; ++sub) {
            const int jcol = j0 + sub * 16 + l16;
            const float bj = slope2 * (float)jcol;
            #pragma unroll
            for (int r = 0; r < 8; ++r) {
                const int irow = mbase + r + 8 * lhalf;
                const float x = s[sub][r] * scl2 + bj;
                sc[sub][r] = (jcol <= irow) ? x : -1e30f;
            }
        }
        _Float16* pw = &Plds[wave * 16 * PPAD];
        #pragma unroll
        for (int r = 0; r < 8; ++r) {
            const float mx   = rowmax16(fmaxf(sc[0][r], sc[1][r]));   // DPP, no LDS
            const float mnew = fmaxf(mrun[r], mx);
            const float corr = exp2f(mrun[r] - mnew);
            mrun[r] = mnew;
            const float p0 = exp2f(sc[0][r] - mnew);
            const float p1 = exp2f(sc[1][r] - mnew);
            const float ps = rowsum16(p0 + p1);                       // DPP, no LDS
            lrun[r] = lrun[r] * corr + ps;
            #pragma unroll
            for (int f = 0; f < 8; ++f) acc[f][r] *= corr;
            const int prow = r + 8 * lhalf;                // D-layout row
            pw[prow * PPAD + l16]      = (_Float16)p0;
            pw[prow * PPAD + 16 + l16] = (_Float16)p1;
        }
        // wave-private LDS transpose: DS ops are in-order; drain counter before reload
        asm volatile("s_wait_dscnt 0" ::: "memory");

        // ---- reload P in A-matrix layout (16x32) ----
        H16 pf;
        {
            const int base = l16 * PPAD + koff;
            pf.q[0] = *(const v4h*)&pw[base + 0];
            pf.q[1] = *(const v4h*)&pw[base + 4];
            pf.q[2] = *(const v4h*)&pw[base + 16];
            pf.q[3] = *(const v4h*)&pw[base + 20];
        }

        // ---- preload ALL V B-fragments, then 8 back-to-back PV WMMAs ----
        H16 vf[8];
        #pragma unroll
        for (int f = 0; f < 8; ++f) {
            const int base = (f * 16 + l16) * VPAD + (lhalf ? 8 : 0);
            #pragma unroll
            for (int g = 0; g < 4; ++g)
                vf[f].q[g] = *(const v4h*)&Vt[cur][base + (g & 1) * 4 + (g >> 1) * 16];
        }
        #pragma unroll
        for (int f = 0; f < 8; ++f)
            acc[f] = __builtin_amdgcn_wmma_f32_16x16x32_f16(
                         false, pf.v, false, vf[f].v, (short)0, acc[f], false, false);

        // ---- convert + store next tile into the alternate buffer ----
        if (more) {
            const int nxt = cur ^ 1;
            #pragma unroll
            for (int it = 0; it < 4; ++it) {
                H4 kh;
                kh.h[0] = (_Float16)kreg[it].x; kh.h[1] = (_Float16)kreg[it].y;
                kh.h[2] = (_Float16)kreg[it].z; kh.h[3] = (_Float16)kreg[it].w;
                *(v4h*)&Klds[nxt][srow[it] * KPAD + scol] = kh.v;
                Vt[nxt][(scol + 0) * VPAD + srow[it]] = (_Float16)vreg[it].x;
                Vt[nxt][(scol + 1) * VPAD + srow[it]] = (_Float16)vreg[it].y;
                Vt[nxt][(scol + 2) * VPAD + srow[it]] = (_Float16)vreg[it].z;
                Vt[nxt][(scol + 3) * VPAD + srow[it]] = (_Float16)vreg[it].w;
            }
        }
        __syncthreads();   // next buffer ready; current buffer free for step+1 writes
    }

    // ---- normalize and store O (fp32) ----
    #pragma unroll
    for (int f = 0; f < 8; ++f) {
        #pragma unroll
        for (int r = 0; r < 8; ++r) {
            const int row = mbase + r + 8 * lhalf;
            out[(bh + (size_t)row) * DHEAD + f * 16 + l16] = acc[f][r] / lrun[r];
        }
    }
}

extern "C" void kernel_launch(void* const* d_in, const int* in_sizes, int n_in,
                              void* d_out, int out_size, void* d_ws, size_t ws_size,
                              hipStream_t stream) {
    const float* q = (const float*)d_in[0];
    const float* k = (const float*)d_in[1];
    const float* v = (const float*)d_in[2];
    float* o = (float*)d_out;
    const int B = in_sizes[0] / (NH * SEQ * DHEAD);
    dim3 grid(SEQ / QTILE, NH, B);
    fa_alibi_kernel<<<grid, 256, 0, stream>>>(q, k, v, o);
}
